// GraphMultisetTransformer_16965120819757
// MI455X (gfx1250) — matrix operand order repaired and checked by reading, not compiled, not run on hardware
//
#include <hip/hip_runtime.h>
#include <hip/hip_bf16.h>
#include <math.h>

typedef _Float16 h8   __attribute__((ext_vector_type(8)));
typedef _Float16 v16h __attribute__((ext_vector_type(16)));
typedef float    v8f  __attribute__((ext_vector_type(8)));

#define BB   16
#define NN   4096
#define CC   256
#define EVD  64
#define HH   8
#define NQ   30
#define NQP  32
#define DH   32

union V16U { v16h v; h8 h[2]; };

__device__ __forceinline__ int regionOf(int n, int d) {
    return (n < d) ? 0 : (n < 2 * d) ? 1 : (n < 3 * d) ? 2 : 3;
}

// ---------------------------------------------------------------------------
// Workspace layout (bytes)
// ---------------------------------------------------------------------------
static constexpr size_t OFF_KD  = 0;                          // B*N*C f16  = 33.5 MB
static constexpr size_t OFF_VT  = OFF_KD  + (size_t)BB*NN*CC*2;   // B*C*N f16 (transposed V)
static constexpr size_t OFF_SC  = OFF_VT  + (size_t)BB*NN*CC*2;   // B*H*32*N f32 = 67 MB (Aw f16 aliased in-place)
static constexpr size_t OFF_WSW = OFF_SC  + (size_t)BB*HH*NQP*NN*4; // 6 swizzled 256x256 f16 = 768 KB
static constexpr size_t OFF_QF  = OFF_WSW + (size_t)6*CC*CC*2;    // B*30*C f32
static constexpr size_t OFF_QH  = OFF_QF  + (size_t)BB*NQ*CC*4;   // B*32*C f16 (padded)
static constexpr size_t OFF_O0  = OFF_QH  + (size_t)BB*NQP*CC*2;  // B*32*C f32

// ---------------------------------------------------------------------------
// 1) Pre-swizzle the 6 weight matrices into WMMA B-matrix lane order:
//    Wsw[mat][kb(8)][cb(16)][lane(32)][e(16)] = W[c][k],
//    c = cb*16 + (lane&15), k = kb*32 + (lane>>4)*16 + e
// ---------------------------------------------------------------------------
__global__ __launch_bounds__(32)
void k_prep_weights(const float* __restrict__ w0, const float* __restrict__ w1,
                    const float* __restrict__ w2, const float* __restrict__ w3,
                    const float* __restrict__ w4, const float* __restrict__ w5,
                    _Float16* __restrict__ Wsw) {
    int mat = blockIdx.x >> 7;
    int rem = blockIdx.x & 127;
    int kb = rem >> 4, cb = rem & 15;
    int lane = threadIdx.x;
    const float* W = (mat == 0) ? w0 : (mat == 1) ? w1 : (mat == 2) ? w2
                   : (mat == 3) ? w3 : (mat == 4) ? w4 : w5;
    int c = cb * 16 + (lane & 15);
    int kbase = kb * 32 + (lane >> 4) * 16;
    _Float16* dst = Wsw + (((size_t)(mat * 8 + kb) * 16 + cb) * 32 + lane) * 16;
#pragma unroll
    for (int e = 0; e < 16; ++e)
        dst[e] = (_Float16)W[(size_t)c * CC + kbase + e];
}

// ---------------------------------------------------------------------------
// 2) Q projection: Q[b,i,c] = ev[b,i,:] . Wq_r[c,:]   (r = i/10)
// ---------------------------------------------------------------------------
__global__ __launch_bounds__(256)
void k_q(const float* __restrict__ ev, const float* __restrict__ WqG,
         const float* __restrict__ WqD, const float* __restrict__ WqL,
         float* __restrict__ Qf, _Float16* __restrict__ Qh) {
    __shared__ float e[NQ * EVD];
    int b = blockIdx.x, c = threadIdx.x;
    for (int idx = c; idx < NQ * EVD; idx += 256)
        e[idx] = ev[(size_t)b * NQ * EVD + idx];
    __syncthreads();
    for (int i = 0; i < NQ; ++i) {
        const float* W = (i < 10) ? WqG : (i < 20) ? WqD : WqL;
        float acc = 0.f;
#pragma unroll 8
        for (int k = 0; k < EVD; ++k) acc += e[i * EVD + k] * W[(size_t)c * EVD + k];
        Qf[((size_t)b * NQ + i) * CC + c]  = acc;
        Qh[((size_t)b * NQP + i) * CC + c] = (_Float16)acc;
    }
    Qh[((size_t)b * NQP + 30) * CC + c] = (_Float16)0.f;
    Qh[((size_t)b * NQP + 31) * CC + c] = (_Float16)0.f;
}

// ---------------------------------------------------------------------------
// 3) Orthogonality penalty wd = (1/B) * sum_{g,h,b} ||G G^T - I||_F
// ---------------------------------------------------------------------------
__global__ __launch_bounds__(512)
void k_wd(const float* __restrict__ Qf, float* __restrict__ wdout) {
    __shared__ float red[512];
    int t = threadIdx.x;
    float fro = 0.f;
    if (t < 384) {
        int g = t / 128, rem = t % 128;
        int h = rem / 16, b = rem % 16;
        float acc = 0.f;
        for (int i = 0; i < 10; ++i) {
            const float* gi = Qf + ((size_t)b * NQ + g * 10 + i) * CC + h * DH;
            for (int j = 0; j < 10; ++j) {
                const float* gj = Qf + ((size_t)b * NQ + g * 10 + j) * CC + h * DH;
                float dot = 0.f;
#pragma unroll 8
                for (int d = 0; d < DH; ++d) dot += gi[d] * gj[d];
                float df = dot - ((i == j) ? 1.f : 0.f);
                acc += df * df;
            }
        }
        fro = sqrtf(acc);
    }
    red[t] = fro;
    __syncthreads();
    for (int s = 256; s > 0; s >>= 1) {
        if (t < s) red[t] += red[t + s];
        __syncthreads();
    }
    if (t == 0) *wdout = red[0] / (float)BB;
}

// ---------------------------------------------------------------------------
// 4) Gated K/V projection with WMMA.  One block = one 16-row n-tile,
//    16 waves cover the 16 c-tiles.  x/ox pulled into LDS with
//    GLOBAL_LOAD_ASYNC_TO_LDS_B128 (ASYNCcnt), then converted f32->f16 in LDS.
// ---------------------------------------------------------------------------
__global__ __launch_bounds__(512)
void k_proj(const float* __restrict__ x, const float* __restrict__ ox,
            const int* __restrict__ numv, const _Float16* __restrict__ Wsw,
            const float* __restrict__ bvG, const float* __restrict__ bvD,
            const float* __restrict__ bvL,
            _Float16* __restrict__ Kd, _Float16* __restrict__ Vt) {
    __shared__ __align__(16) float    fx[16 * CC];   // raw f32 tile of x
    __shared__ __align__(16) float    fo[16 * CC];   // raw f32 tile of origin_x
    __shared__ __align__(16) _Float16 lx[16 * CC];
    __shared__ __align__(16) _Float16 lo[16 * CC];
    int b = blockIdx.x >> 8;
    int nt = blockIdx.x & 255;
    int nbase = nt * 16;
    int d = numv[b + 1] - numv[b];

    // ---- async global -> LDS staging (32 bytes per thread per tensor) ----
    {
        unsigned gOff  = (unsigned)((((size_t)b * NN + nbase) * CC) * 4) + threadIdx.x * 32u;
        unsigned lOffX = (unsigned)(uintptr_t)&fx[0] + threadIdx.x * 32u;
        unsigned lOffO = (unsigned)(uintptr_t)&fo[0] + threadIdx.x * 32u;
        asm volatile("global_load_async_to_lds_b128 %0, %1, %2"
                     :: "v"(lOffX), "v"(gOff), "s"(x) : "memory");
        asm volatile("global_load_async_to_lds_b128 %0, %1, %2 offset:16"
                     :: "v"(lOffX), "v"(gOff), "s"(x) : "memory");
        asm volatile("global_load_async_to_lds_b128 %0, %1, %2"
                     :: "v"(lOffO), "v"(gOff), "s"(ox) : "memory");
        asm volatile("global_load_async_to_lds_b128 %0, %1, %2 offset:16"
                     :: "v"(lOffO), "v"(gOff), "s"(ox) : "memory");
        asm volatile("s_wait_asynccnt 0" ::: "memory");
    }
    __syncthreads();
    // ---- f32 -> f16 conversion pass in LDS ----
#pragma unroll
    for (int u = 0; u < 8; ++u) {
        int idx = threadIdx.x + u * 512;
        lx[idx] = (_Float16)fx[idx];
        lo[idx] = (_Float16)fo[idx];
    }
    __syncthreads();

    int cb = threadIdx.x >> 5;          // c-tile handled by this wave
    int lane = threadIdx.x & 31;
    int khalf = lane >> 4;
    int mA = lane & 15;                 // A-matrix row of this lane
    int c = cb * 16 + (lane & 15);      // output column of this lane
    int regA = regionOf(nbase + mA, d);

    v8f ck = {}; v8f cv = {};
    v16h zero = {};
    int rF = regionOf(nbase, d);
    int rL = regionOf(nbase + 15, d); if (rL > 2) rL = 2;

    for (int r = rF; r <= rL; ++r) {
#pragma unroll
        for (int kb = 0; kb < 8; ++kb) {
            int k0 = kb * 32 + khalf * 16;
            V16U ax, ao, bk, bv;
            ax.h[0] = *(const h8*)&lx[mA * CC + k0];
            ax.h[1] = *(const h8*)&lx[mA * CC + k0 + 8];
            ao.h[0] = *(const h8*)&lo[mA * CC + k0];
            ao.h[1] = *(const h8*)&lo[mA * CC + k0 + 8];
            v16h axm = (regA == r) ? ax.v : zero;
            v16h aom = (regA == r) ? ao.v : zero;
            const _Float16* bkp = Wsw + ((((size_t)r       * 8 + kb) * 16 + cb) * 32 + lane) * 16;
            const _Float16* bvp = Wsw + ((((size_t)(3 + r) * 8 + kb) * 16 + cb) * 32 + lane) * 16;
            bk.h[0] = *(const h8*)bkp;       bk.h[1] = *(const h8*)(bkp + 8);
            bv.h[0] = *(const h8*)bvp;       bv.h[1] = *(const h8*)(bvp + 8);
            ck = __builtin_amdgcn_wmma_f32_16x16x32_f16(false, axm, false, bk.v, (short)0, ck, false, false);
            cv = __builtin_amdgcn_wmma_f32_16x16x32_f16(false, aom, false, bv.v, (short)0, cv, false, false);
        }
    }

    float bv0 = bvG[c], bv1 = bvD[c], bv2 = bvL[c];
#pragma unroll
    for (int v = 0; v < 8; ++v) {
        int m = v + 8 * khalf;                  // C/D row of VGPR v for this lane
        int rr = regionOf(nbase + m, d);
        float bias = (rr == 0) ? bv0 : (rr == 1) ? bv1 : (rr == 2) ? bv2 : 0.f;
        Kd[((size_t)b * NN + nbase + m) * CC + c]  = (_Float16)ck[v];
        Vt[((size_t)b * CC + c) * NN + nbase + m]  = (_Float16)(cv[v] + bias);
    }
}

// ---------------------------------------------------------------------------
// 5) scores = (Q K^T)/1.6 + mask  (one WMMA per 16q x 16k tile, dh = 32 = K)
// ---------------------------------------------------------------------------
__global__ __launch_bounds__(256)
void k_qk(const _Float16* __restrict__ Qh, const _Float16* __restrict__ Kd,
          const float* __restrict__ mask, float* __restrict__ scores) {
    int wg = (blockIdx.x << 3) + (threadIdx.x >> 5);
    int lane = threadIdx.x & 31;
    int kt = wg & 255;
    int qt = (wg >> 8) & 1;
    int bh = wg >> 9;
    int b = bh >> 3, h = bh & 7;
    int khalf = lane >> 4, mrow = lane & 15;

    int q = qt * 16 + mrow;
    const _Float16* qrow = Qh + ((size_t)b * NQP + q) * CC + h * DH;
    V16U a, bm;
    a.h[0] = *(const h8*)(qrow + khalf * 16);
    a.h[1] = *(const h8*)(qrow + khalf * 16 + 8);

    int kpos = kt * 16 + mrow;
    const _Float16* krow = Kd + ((size_t)b * NN + kpos) * CC + h * DH + khalf * 16;
    bm.h[0] = *(const h8*)krow;
    bm.h[1] = *(const h8*)(krow + 8);

    v8f cz = {};
    v8f cr = __builtin_amdgcn_wmma_f32_16x16x32_f16(false, a.v, false, bm.v, (short)0, cz, false, false);

    int ncol = kt * 16 + mrow;
#pragma unroll
    for (int v = 0; v < 8; ++v) {
        int qq = qt * 16 + v + 8 * khalf;
        int qc = (qq < NQ) ? qq : (NQ - 1);
        float val = cr[v] * 0.625f + mask[((size_t)b * NQ + qc) * NN + ncol];
        scores[((size_t)bh * NQP + qq) * NN + ncol] = val;
    }
}

// ---------------------------------------------------------------------------
// 6) Row softmax over k (N=4096); write Aw f16 in-place over own score row.
// ---------------------------------------------------------------------------
__global__ __launch_bounds__(256)
void k_softmax(float* __restrict__ scores) {
    __shared__ float srow[NN];
    __shared__ float red[256];
    size_t row = blockIdx.x;
    float* sp = scores + row * NN;
    int t = threadIdx.x;
    for (int k = t; k < NN; k += 256) srow[k] = sp[k];
    __syncthreads();
    float mx = -3.0e38f;
    for (int k = t; k < NN; k += 256) mx = fmaxf(mx, srow[k]);
    red[t] = mx; __syncthreads();
    for (int s = 128; s > 0; s >>= 1) { if (t < s) red[t] = fmaxf(red[t], red[t + s]); __syncthreads(); }
    mx = red[0]; __syncthreads();
    float sm = 0.f;
    for (int k = t; k < NN; k += 256) sm += __expf(srow[k] - mx);
    red[t] = sm; __syncthreads();
    for (int s = 128; s > 0; s >>= 1) { if (t < s) red[t] += red[t + s]; __syncthreads(); }
    float inv = 1.0f / red[0];
    _Float16* aw = (_Float16*)sp;      // safe: this block owns the whole row, staged in LDS
    for (int k = t; k < NN; k += 256) aw[k] = (_Float16)(__expf(srow[k] - mx) * inv);
}

// ---------------------------------------------------------------------------
// 7) out0 = Aw @ V   (contraction over k = 4096, 128 WMMA steps per wave)
// ---------------------------------------------------------------------------
__global__ __launch_bounds__(256)
void k_av(const float* __restrict__ scores, const _Float16* __restrict__ Vt,
          float* __restrict__ out0) {
    int wg = (blockIdx.x << 3) + (threadIdx.x >> 5);   // 0..511
    int lane = threadIdx.x & 31;
    int dt = wg & 1;
    int qt = (wg >> 1) & 1;
    int bh = wg >> 2;
    int b = bh >> 3, h = bh & 7;
    int khalf = lane >> 4, mrow = lane & 15;

    const _Float16* awBase = (const _Float16*)scores;
    size_t rowHalf = ((size_t)bh * NQP + qt * 16 + mrow) * (size_t)(NN * 2);
    int dcol = h * DH + dt * 16 + mrow;
    const _Float16* vcol = Vt + ((size_t)b * CC + dcol) * NN;

    v8f acc = {};
    for (int kb = 0; kb < NN / 32; ++kb) {
        int kbase = kb * 32;
        V16U a, bm;
        a.h[0]  = *(const h8*)(awBase + rowHalf + kbase + khalf * 16);
        a.h[1]  = *(const h8*)(awBase + rowHalf + kbase + khalf * 16 + 8);
        bm.h[0] = *(const h8*)(vcol + kbase + khalf * 16);
        bm.h[1] = *(const h8*)(vcol + kbase + khalf * 16 + 8);
        acc = __builtin_amdgcn_wmma_f32_16x16x32_f16(false, a.v, false, bm.v, (short)0, acc, false, false);
    }
#pragma unroll
    for (int v = 0; v < 8; ++v) {
        int q = qt * 16 + v + 8 * khalf;
        out0[((size_t)b * NQP + q) * CC + dcol] = acc[v];
    }
}

// ---------------------------------------------------------------------------
// 8) Epilogue: out = out0 + relu(out0 @ Wo^T + bo); L2-normalize each row.
// ---------------------------------------------------------------------------
__global__ __launch_bounds__(256)
void k_epilogue(const float* __restrict__ out0, const float* __restrict__ Wo,
                const float* __restrict__ bo, float* __restrict__ out) {
    __shared__ float row[CC];
    __shared__ float red[256];
    int b = blockIdx.x / NQ, q = blockIdx.x % NQ;
    int c = threadIdx.x;
    row[c] = out0[((size_t)b * NQP + q) * CC + c];
    __syncthreads();
    float y = bo[c];
    const float* w = Wo + (size_t)c * CC;
#pragma unroll 8
    for (int j = 0; j < CC; ++j) y += row[j] * w[j];
    float r = row[c] + fmaxf(y, 0.f);
    red[c] = r * r;
    __syncthreads();
    for (int s = 128; s > 0; s >>= 1) { if (c < s) red[c] += red[c + s]; __syncthreads(); }
    float denom = fmaxf(sqrtf(red[0]), 1e-12f);
    out[((size_t)b * NQ + q) * CC + c] = r / denom;
}

// ---------------------------------------------------------------------------
extern "C" void kernel_launch(void* const* d_in, const int* in_sizes, int n_in,
                              void* d_out, int out_size, void* d_ws, size_t ws_size,
                              hipStream_t stream) {
    const float* x    = (const float*)d_in[0];
    const float* ox   = (const float*)d_in[1];
    const float* ev   = (const float*)d_in[2];
    const float* mask = (const float*)d_in[3];
    const int*   numv = (const int*)d_in[4];
    const float* WqG = (const float*)d_in[5];
    const float* WqD = (const float*)d_in[6];
    const float* WqL = (const float*)d_in[7];
    const float* WkG = (const float*)d_in[8];
    const float* WkD = (const float*)d_in[9];
    const float* WkL = (const float*)d_in[10];
    const float* WvG = (const float*)d_in[11];
    const float* bvG = (const float*)d_in[12];
    const float* WvD = (const float*)d_in[13];
    const float* bvD = (const float*)d_in[14];
    const float* WvL = (const float*)d_in[15];
    const float* bvL = (const float*)d_in[16];
    const float* Wo  = (const float*)d_in[17];
    const float* bo  = (const float*)d_in[18];

    char* ws = (char*)d_ws;
    _Float16* Kd  = (_Float16*)(ws + OFF_KD);
    _Float16* Vt  = (_Float16*)(ws + OFF_VT);
    float*    sc  = (float*)   (ws + OFF_SC);
    _Float16* Wsw = (_Float16*)(ws + OFF_WSW);
    float*    Qf  = (float*)   (ws + OFF_QF);
    _Float16* Qh  = (_Float16*)(ws + OFF_QH);
    float*    o0  = (float*)   (ws + OFF_O0);

    float* out = (float*)d_out;
    float* wd  = out + (size_t)BB * NQ * CC;   // element 122880

    k_prep_weights<<<6 * 128, 32, 0, stream>>>(WkG, WkD, WkL, WvG, WvD, WvL, Wsw);
    k_q<<<BB, 256, 0, stream>>>(ev, WqG, WqD, WqL, Qf, Qh);
    k_wd<<<1, 512, 0, stream>>>(Qf, wd);
    k_proj<<<BB * (NN / 16), 512, 0, stream>>>(x, ox, numv, Wsw, bvG, bvD, bvL, Kd, Vt);
    k_qk<<<(BB * HH * 2 * (NN / 16)) / 8, 256, 0, stream>>>(Qh, Kd, mask, sc);
    k_softmax<<<BB * HH * NQP, 256, 0, stream>>>(sc);
    k_av<<<(BB * HH * 2 * 2) / 8, 256, 0, stream>>>(sc, Vt, o0);
    k_epilogue<<<BB * NQ, 256, 0, stream>>>(o0, Wo, bo, out);
}
